// MaskCriterion_88596585382237
// MI455X (gfx1250) — compile-verified, matrix operand order checked by reading, not run on hardware
//
#include <hip/hip_runtime.h>

typedef __attribute__((ext_vector_type(16))) _Float16 v16h;
typedef __attribute__((ext_vector_type(8)))  float    v8f;

#define B_DIM 4096
#define D_DIM 8192
#define ROWS_PER_BLOCK 16
#define COL_CHUNKS 4
#define CHUNK_COLS (D_DIM / COL_CHUNKS)   // 2048
#define STEP_COLS 256                      // 8 waves * 32 cols
#define NSTEPS (CHUNK_COLS / STEP_COLS)    // 8

__device__ __forceinline__ float smooth_l1(float x, float y) {
    float d  = x - y;
    float ad = __builtin_fabsf(d);
    return ad < 1.0f ? 0.5f * d * d : ad - 0.5f;
}

__global__ void zero_out_kernel(float* __restrict__ out) { out[0] = 0.0f; }

__global__ void __launch_bounds__(256)
mask_criterion_kernel(const float* __restrict__ inp,
                      const float* __restrict__ tgt,
                      const int*   __restrict__ mask,
                      float*       __restrict__ out) {
    __shared__ float rowacc[ROWS_PER_BLOCK];
    __shared__ float blocksum;

    const int tid  = threadIdx.x;
    const int wave = tid >> 5;
    const int lane = tid & 31;
    const int mrow = lane & 15;     // matrix row this lane carries (A layout)
    const int hi   = lane >> 4;     // lane group: 0 -> K {0..7,16..23}, 1 -> +8

    const int  rowg = blockIdx.x;                     // 0..255
    const long row  = (long)rowg * ROWS_PER_BLOCK + mrow;
    const long rb   = row * (long)D_DIM;
    const int  col0 = blockIdx.y * CHUNK_COLS + wave * 32 + hi * 8;

    if (tid < ROWS_PER_BLOCK) rowacc[tid] = 0.0f;
    if (tid == 0) blocksum = 0.0f;
    __syncthreads();

    v16h bones;
#pragma unroll
    for (int i = 0; i < 16; ++i) bones[i] = (_Float16)1.0f;

    v8f acc = {};

#pragma unroll 1
    for (int step = 0; step < NSTEPS; ++step) {
        const float* ip = inp + rb + col0 + step * STEP_COLS;
        const float* tp = tgt + rb + col0 + step * STEP_COLS;

        if (step + 1 < NSTEPS) {
            __builtin_prefetch(ip + STEP_COLS, 0, 0);
            __builtin_prefetch(tp + STEP_COLS, 0, 0);
        }

        float4 a0 = *(const float4*)(ip + 0);
        float4 a1 = *(const float4*)(ip + 4);
        float4 a2 = *(const float4*)(ip + 16);
        float4 a3 = *(const float4*)(ip + 20);
        float4 t0 = *(const float4*)(tp + 0);
        float4 t1 = *(const float4*)(tp + 4);
        float4 t2 = *(const float4*)(tp + 16);
        float4 t3 = *(const float4*)(tp + 20);

        v16h a;
        a[0]  = (_Float16)smooth_l1(a0.x, t0.x);
        a[1]  = (_Float16)smooth_l1(a0.y, t0.y);
        a[2]  = (_Float16)smooth_l1(a0.z, t0.z);
        a[3]  = (_Float16)smooth_l1(a0.w, t0.w);
        a[4]  = (_Float16)smooth_l1(a1.x, t1.x);
        a[5]  = (_Float16)smooth_l1(a1.y, t1.y);
        a[6]  = (_Float16)smooth_l1(a1.z, t1.z);
        a[7]  = (_Float16)smooth_l1(a1.w, t1.w);
        a[8]  = (_Float16)smooth_l1(a2.x, t2.x);
        a[9]  = (_Float16)smooth_l1(a2.y, t2.y);
        a[10] = (_Float16)smooth_l1(a2.z, t2.z);
        a[11] = (_Float16)smooth_l1(a2.w, t2.w);
        a[12] = (_Float16)smooth_l1(a3.x, t3.x);
        a[13] = (_Float16)smooth_l1(a3.y, t3.y);
        a[14] = (_Float16)smooth_l1(a3.z, t3.z);
        a[15] = (_Float16)smooth_l1(a3.w, t3.w);

        // D = elem(16x32) * ones(32x16) + C  -> per-row sums in every column of C
        acc = __builtin_amdgcn_wmma_f32_16x16x32_f16(
            /*neg_a=*/false, a, /*neg_b=*/false, bones,
            /*c_mod=*/(short)0, acc, /*reuse_a=*/false, /*reuse_b=*/false);
    }

    // C layout: lane 0 holds M=0..7 at N=0 in acc[0..7]; lane 16 holds M=8..15.
    if (lane == 0 || lane == 16) {
#pragma unroll
        for (int i = 0; i < 8; ++i)
            atomicAdd(&rowacc[hi * 8 + i], acc[i]);
    }
    __syncthreads();

    if (tid < ROWS_PER_BLOCK) {
        float m = (float)mask[rowg * ROWS_PER_BLOCK + tid];   // mask in {0,1}
        float v = rowacc[tid] * m * (1.0f / (float)D_DIM);
        atomicAdd(&blocksum, v);
    }
    __syncthreads();

    if (tid == 0) atomicAdd(out, blocksum);
}

extern "C" void kernel_launch(void* const* d_in, const int* in_sizes, int n_in,
                              void* d_out, int out_size, void* d_ws, size_t ws_size,
                              hipStream_t stream) {
    const float* inp  = (const float*)d_in[0];
    const float* tgt  = (const float*)d_in[1];
    const int*   mask = (const int*)d_in[2];
    float*       out  = (float*)d_out;

    zero_out_kernel<<<1, 1, 0, stream>>>(out);

    dim3 grid(B_DIM / ROWS_PER_BLOCK, COL_CHUNKS);  // 256 x 4 = 1024 blocks
    mask_criterion_kernel<<<grid, 256, 0, stream>>>(inp, tgt, mask, out);
}
